// GraphOUProcessDebug_71854802862394
// MI455X (gfx1250) — 8-bit matrix operand order repaired and checked by reading, compiled, not run on hardware
//
#include <hip/hip_runtime.h>
#include <hip/hip_bf16.h>
#include <stdint.h>

#define G_    2
#define BS_   64
#define N_    512
#define T_    200
#define NCHAIN (G_ * BS_)

#define ROWP  528
#define AT_LDS_BYTES (N_ * ROWP)
#define SMEM_SBUF (AT_LDS_BYTES)
#define SMEM_XF32 (SMEM_SBUF + 2048)
#define SMEM_XHI  (SMEM_XF32 + 2048)
#define SMEM_XLO  (SMEM_XHI + 512)
#define SMEM_TOTAL (SMEM_XLO + 512)

typedef __attribute__((ext_vector_type(16))) int   v16i;
typedef __attribute__((ext_vector_type(8)))  float v8f;

#define MRS 0.02f
#define LVL 0.5f
#define VARF 0.16f

__device__ __forceinline__ unsigned enc_fp8(float f) {
  unsigned bits = __float_as_uint(f);
  unsigned sgn  = (bits >> 24) & 0x80u;
  float af = fabsf(f);
  if (af >= 448.0f) return sgn | 0x7Eu;
  if (af < 0x1p-6f) {
    int m = (int)__builtin_rintf(af * 512.0f);
    if (m >= 8) return sgn | 0x08u;
    return sgn | (unsigned)m;
  }
  unsigned ab = __float_as_uint(af) + (1u << 19);
  int e = (int)((ab >> 23) & 0xFFu) - 127;
  if (e > 8) return sgn | 0x7Eu;
  return sgn | ((unsigned)(e + 7) << 3) | ((ab >> 20) & 7u);
}
__device__ __forceinline__ float dec_fp8(unsigned v) {
  unsigned e = (v >> 3) & 0xFu, m = v & 7u;
  float mag = (e == 0u) ? (float)m * 0x1p-9f
                        : __uint_as_float(((e + 120u) << 23) | (m << 20));
  return (v & 0x80u) ? -mag : mag;
}

__global__ void prep_at_kernel(const float* __restrict__ adj,
                               unsigned char* __restrict__ at) {
  __shared__ float tile[32][33];
  int id = blockIdx.x;
  int ti = id & 15; id >>= 4;
  int tj = id & 15; id >>= 4;
  int g  = id & 1;  id >>= 1;
  int b  = id;
  const float* src = adj + (((size_t)(g * BS_ + b)) * N_ + (size_t)tj * 32) * N_ + ti * 32;
  int tx = threadIdx.x, ty = threadIdx.y;
  #pragma unroll
  for (int rr = 0; rr < 32; rr += 8)
    tile[rr + ty][tx] = src[(size_t)(rr + ty) * N_ + tx];
  __syncthreads();
  unsigned char* dst = at + ((size_t)(b * G_ + g)) * N_ * N_;
  int tid = ty * 32 + tx;
  int i_l = tid >> 3;
  int j4  = tid & 7;
  unsigned v = 0;
  #pragma unroll
  for (int k = 0; k < 4; ++k) {
    float a = tile[j4 * 4 + k][i_l];
    v |= (a != 0.0f ? 0x38u : 0u) << (8 * k);
  }
  *(unsigned*)(dst + (size_t)(ti * 32 + i_l) * N_ + tj * 32 + j4 * 4) = v;
}

__global__ void prep_deg_kernel(const float* __restrict__ adj,
                                float* __restrict__ deg) {
  int g = blockIdx.x, i = threadIdx.x;
  const float* src = adj + ((size_t)(g * BS_)) * N_ * N_;
  float s = 0.f;
  for (int j = 0; j < N_; ++j) s += src[(size_t)j * N_ + i];
  deg[g * N_ + i] = s;
}

__global__ __launch_bounds__(512)
void ou_main_kernel(const unsigned char* __restrict__ at,
                    const float* __restrict__ deg,
                    const float* __restrict__ times,
                    const int*   __restrict__ skills,
                    const float* __restrict__ noise,
                    float* __restrict__ out) {
  extern __shared__ char smem[];
  unsigned char* AT   = (unsigned char*)smem;
  float*         sbuf = (float*)(smem + SMEM_SBUF);
  float*         xf   = (float*)(smem + SMEM_XF32);
  unsigned char* xhi  = (unsigned char*)(smem + SMEM_XHI);
  unsigned char* xlo  = (unsigned char*)(smem + SMEM_XLO);

  const int chain = blockIdx.x;
  const int b = chain >> 1, g = chain & 1;
  const int tid  = threadIdx.x;
  const int lane = tid & 31;
  const int wave = tid >> 5;
  const int hh   = lane >> 4;
  const int lrow = lane & 15;

  const unsigned char* at_g = at + (size_t)chain * (N_ * N_);

  {
    unsigned ldsbase = (unsigned)(size_t)(void*)AT;
    for (int it = 0; it < 32; ++it) {
      int idx = it * 512 + tid;
      int row = idx >> 5, c = (idx & 31) * 16;
      unsigned long long ga = (unsigned long long)(size_t)(at_g + (size_t)row * N_ + c);
      unsigned la = ldsbase + (unsigned)(row * ROWP + c);
      asm volatile("global_load_async_to_lds_b128 %0, %1, off"
                   :: "v"(la), "v"(ga) : "memory");
    }
    asm volatile("s_wait_asynccnt 0" ::: "memory");
  }

  xf[tid] = 0.0f; xhi[tid] = 0; xlo[tid] = 0;
  const float dgv  = deg[g * N_ + tid];
  const float invd = 1.0f / (dgv + 1e-7f);
  const bool  zm   = (dgv == 0.0f);
  __syncthreads();

  const unsigned char* rowA0 = AT + (size_t)(32 * wave + lrow) * ROWP + 16 * hh;
  const unsigned char* rowA1 = rowA0 + 16 * ROWP;

  float t_prev = 0.0f;
  for (int t = 0; t < T_; ++t) {
    v8f dh0 = {}, dl0 = {}, dh1 = {}, dl1 = {};
    #pragma unroll
    for (int kc = 0; kc < 4; ++kc) {
      const int kb = kc * 128;
      const int boff = kb + 16 * hh;
      int4 h0 = *(const int4*)(xhi + boff +  0);
      int4 h1 = *(const int4*)(xhi + boff + 32);
      int4 h2 = *(const int4*)(xhi + boff + 64);
      int4 h3 = *(const int4*)(xhi + boff + 96);
      int4 l0 = *(const int4*)(xlo + boff +  0);
      int4 l1 = *(const int4*)(xlo + boff + 32);
      int4 l2 = *(const int4*)(xlo + boff + 64);
      int4 l3 = *(const int4*)(xlo + boff + 96);
      v16i bh, bl;
      bh[0]=h0.x; bh[1]=h0.y; bh[2]=h0.z; bh[3]=h0.w;
      bh[4]=h1.x; bh[5]=h1.y; bh[6]=h1.z; bh[7]=h1.w;
      bh[8]=h2.x; bh[9]=h2.y; bh[10]=h2.z; bh[11]=h2.w;
      bh[12]=h3.x; bh[13]=h3.y; bh[14]=h3.z; bh[15]=h3.w;
      bl[0]=l0.x; bl[1]=l0.y; bl[2]=l0.z; bl[3]=l0.w;
      bl[4]=l1.x; bl[5]=l1.y; bl[6]=l1.z; bl[7]=l1.w;
      bl[8]=l2.x; bl[9]=l2.y; bl[10]=l2.z; bl[11]=l2.w;
      bl[12]=l3.x; bl[13]=l3.y; bl[14]=l3.z; bl[15]=l3.w;
      {
        const unsigned char* ar = rowA0 + kb;
        int2 a0 = *(const int2*)(ar +   0), a1 = *(const int2*)(ar +   8);
        int2 a2 = *(const int2*)(ar +  32), a3 = *(const int2*)(ar +  40);
        int2 a4 = *(const int2*)(ar +  64), a5 = *(const int2*)(ar +  72);
        int2 a6 = *(const int2*)(ar +  96), a7 = *(const int2*)(ar + 104);
        v16i av;
        av[0]=a0.x; av[1]=a0.y; av[2]=a1.x; av[3]=a1.y;
        av[4]=a2.x; av[5]=a2.y; av[6]=a3.x; av[7]=a3.y;
        av[8]=a4.x; av[9]=a4.y; av[10]=a5.x; av[11]=a5.y;
        av[12]=a6.x; av[13]=a6.y; av[14]=a7.x; av[15]=a7.y;
        dh0 = __builtin_amdgcn_wmma_f32_16x16x128_fp8_fp8(av, bh, (short)0, dh0, false, false);
        dl0 = __builtin_amdgcn_wmma_f32_16x16x128_fp8_fp8(av, bl, (short)0, dl0, false, false);
      }
      {
        const unsigned char* ar = rowA1 + kb;
        int2 a0 = *(const int2*)(ar +   0), a1 = *(const int2*)(ar +   8);
        int2 a2 = *(const int2*)(ar +  32), a3 = *(const int2*)(ar +  40);
        int2 a4 = *(const int2*)(ar +  64), a5 = *(const int2*)(ar +  72);
        int2 a6 = *(const int2*)(ar +  96), a7 = *(const int2*)(ar + 104);
        v16i av;
        av[0]=a0.x; av[1]=a0.y; av[2]=a1.x; av[3]=a1.y;
        av[4]=a2.x; av[5]=a2.y; av[6]=a3.x; av[7]=a3.y;
        av[8]=a4.x; av[9]=a4.y; av[10]=a5.x; av[11]=a5.y;
        av[12]=a6.x; av[13]=a6.y; av[14]=a7.x; av[15]=a7.y;
        dh1 = __builtin_amdgcn_wmma_f32_16x16x128_fp8_fp8(av, bh, (short)0, dh1, false, false);
        dl1 = __builtin_amdgcn_wmma_f32_16x16x128_fp8_fp8(av, bl, (short)0, dl1, false, false);
      }
    }
    if (lrow == 0) {
      const int base = 32 * wave + 8 * hh;
      #pragma unroll
      for (int r = 0; r < 8; ++r) sbuf[base + r]      = dh0[r] + 0.0625f * dl0[r];
      #pragma unroll
      for (int r = 0; r < 8; ++r) sbuf[base + 16 + r] = dh1[r] + 0.0625f * dl1[r];
    }
    __syncthreads();

    float tc = times[b * T_ + t];
    float dt = tc - t_prev; t_prev = tc;
    float e  = __expf(-MRS * dt);
    float sc = __builtin_sqrtf(VARF * (1.0f - e * e));
    float s  = sbuf[tid] * invd;
    if (zm) s = 1.0f;
    float xo = xf[tid];
    float mean = xo * e + (1.0f - e) * (LVL * s);
    float xn = noise[(((size_t)t * BS_ + b) * G_ + g) * N_ + tid] * sc + mean;
    xf[tid] = xn;
    unsigned h8 = enc_fp8(xn);
    float    rh = dec_fp8(h8);
    unsigned l8 = enc_fp8((xn - rh) * 16.0f);
    xhi[tid] = (unsigned char)h8;
    xlo[tid] = (unsigned char)l8;
    int sk = skills[b * T_ + t];
    if (tid == sk)
      out[((size_t)t * BS_ + b) * G_ + g] = 1.0f / (1.0f + __expf(-xn));
    __syncthreads();
  }
}

extern "C" void kernel_launch(void* const* d_in, const int* in_sizes, int n_in,
                              void* d_out, int out_size, void* d_ws, size_t ws_size,
                              hipStream_t stream) {
  const float* adj    = (const float*)d_in[0];
  const float* times  = (const float*)d_in[1];
  const int*   skills = (const int*)d_in[2];
  const float* noise  = (const float*)d_in[3];
  float* out = (float*)d_out;

  unsigned char* at_ws  = (unsigned char*)d_ws;
  float*         deg_ws = (float*)((char*)d_ws + (size_t)NCHAIN * N_ * N_);

  (void)in_sizes; (void)n_in; (void)out_size; (void)ws_size;

  hipFuncSetAttribute((const void*)ou_main_kernel,
                      hipFuncAttributeMaxDynamicSharedMemorySize, SMEM_TOTAL);

  prep_at_kernel<<<dim3(BS_ * G_ * 16 * 16), dim3(32, 8), 0, stream>>>(adj, at_ws);
  prep_deg_kernel<<<dim3(G_), dim3(N_), 0, stream>>>(adj, deg_ws);
  ou_main_kernel<<<dim3(NCHAIN), dim3(512), SMEM_TOTAL, stream>>>(
      at_ws, deg_ws, times, skills, noise, out);
}
